// LRSAttention_72232759984876
// MI455X (gfx1250) — matrix operand order repaired and checked by reading, not compiled, not run on hardware
//
#include <hip/hip_runtime.h>

// ---------------------------------------------------------------------------
// LRS attention for MI455X (gfx1250, wave32, WMMA).
// Math: softmax(-(a - 2qk + b)/den) == softmax((2/den) * qk)  (a,b const per head)
// so scale = 2*ALPHA / (||q||_F * ||x^T||_inf) and this is plain scaled attention.
// All GEMMs use v_wmma_f32_16x16x32_bf16 (fp32 accumulate).
// Tile staging: gfx1250 async global->LDS DMA (ASYNCcnt), double-buffered so
// the next tile's DMA overlaps the current tile's WMMA work. Async loads
// complete in order, so with 4 copies/tile/wave, "s_wait_asynccnt 4" after
// issuing the next tile guarantees the current tile has landed.
// ---------------------------------------------------------------------------

#define B_   2
#define N_   2048
#define D_   1024
#define H_   16
#define HD_  64
#define ALPHA_ 1.0f

typedef __attribute__((ext_vector_type(16))) __bf16 v16bf;
typedef __attribute__((ext_vector_type(8)))  float  v8f;

union FragA { v16bf v; uint4 q[2]; };

__device__ __forceinline__ __bf16 f2bf(float f) {
  union { float f; unsigned u; } in; in.f = f;
  unsigned u = in.u + (0x7FFFu + ((in.u >> 16) & 1u)); // round-to-nearest-even
  union { unsigned short s; __bf16 b; } out; out.s = (unsigned short)(u >> 16);
  return out.b;
}
__device__ __forceinline__ float bf2f(__bf16 b) {
  union { __bf16 b; unsigned short s; } in; in.b = b;
  union { unsigned u; float f; } out; out.u = ((unsigned)in.s) << 16;
  return out.f;
}
__device__ __forceinline__ v8f wmma_bf16(FragA a, FragA b, v8f c) {
  // D = A(16x32 bf16) * B(32x16 bf16) + C(16x16 f32)
  return __builtin_amdgcn_wmma_f32_16x16x32_bf16(
      /*neg_a=*/false, a.v, /*neg_b=*/false, b.v,
      /*c_mod=*/(short)0, c, /*reuse_a=*/false, /*reuse_b=*/false);
}

// 16B async DMA: global -> LDS, tracked by ASYNCcnt (no VGPR staging).
// Generic pointer to __shared__ truncated to 32 bits == wave LDS byte offset.
__device__ __forceinline__ void async_copy16(void* lds_dst, const void* gsrc) {
  unsigned lds = (unsigned)(uintptr_t)lds_dst;
  asm volatile("global_load_async_to_lds_b128 %0, %1, off"
               :: "v"(lds), "v"(gsrc)
               : "memory");
}
__device__ __forceinline__ void async_wait0() {
  asm volatile("s_wait_asynccnt 0x0" ::: "memory");
}
__device__ __forceinline__ void async_wait4() {   // previous tile (4 copies) done
  asm volatile("s_wait_asynccnt 0x4" ::: "memory");
}

// ------------------------- small prep kernels ------------------------------

__global__ __launch_bounds__(256) void f32_to_bf16_kernel(
    const float* __restrict__ in, __bf16* __restrict__ out, int n) {
  int i = blockIdx.x * 256 + threadIdx.x;
  if (i < n) out[i] = f2bf(in[i]);
}

// out[C][R] (bf16) = transpose of in[R][C] (f32)
__global__ __launch_bounds__(256) void transpose_f32_to_bf16_kernel(
    const float* __restrict__ in, __bf16* __restrict__ out, int R, int C) {
  int o = blockIdx.x * 256 + threadIdx.x;
  if (o < R * C) {
    int c = o / R, r = o % R;
    out[o] = f2bf(in[(size_t)r * C + c]);
  }
}

// colabs[b][d] = sum_n |x[b,n,d]|
__global__ __launch_bounds__(256) void colabs_kernel(
    const float* __restrict__ x, float* __restrict__ colabs) {
  int d = blockIdx.x * 256 + threadIdx.x;   // 0..1023
  int b = blockIdx.y;                       // 0..1
  float s = 0.f;
  for (int n = 0; n < N_; ++n) s += fabsf(x[((size_t)b * N_ + n) * D_ + d]);
  colabs[b * D_ + d] = s;
}

// infxt[b] = max_d colabs[b][d]
__global__ __launch_bounds__(256) void infmax_kernel(
    const float* __restrict__ colabs, float* __restrict__ infxt) {
  __shared__ float red[256];
  int b = blockIdx.x, tid = threadIdx.x;
  float m = 0.f;
  for (int i = tid; i < D_; i += 256) m = fmaxf(m, colabs[b * D_ + i]);
  red[tid] = m; __syncthreads();
  for (int s = 128; s > 0; s >>= 1) {
    if (tid < s) red[tid] = fmaxf(red[tid], red[tid + s]);
    __syncthreads();
  }
  if (tid == 0) infxt[b] = red[0];
}

// froq[bh] = sum over (n,hd) of q^2   (deterministic tree reduction)
__global__ __launch_bounds__(256) void froq_kernel(
    const __bf16* __restrict__ qb, float* __restrict__ froq) {
  __shared__ float red[256];
  int bh = blockIdx.x, tid = threadIdx.x;
  const __bf16* p = qb + (size_t)bh * N_ * HD_;
  float s = 0.f;
  for (int i = tid; i < N_ * HD_; i += 256) { float v = bf2f(p[i]); s += v * v; }
  red[tid] = s; __syncthreads();
  for (int st = 128; st > 0; st >>= 1) {
    if (tid < st) red[tid] += red[tid + st];
    __syncthreads();
  }
  if (tid == 0) froq[bh] = red[0];
}

__global__ __launch_bounds__(32) void scale_kernel(
    const float* __restrict__ froq, const float* __restrict__ infxt,
    float* __restrict__ scale) {
  int i = threadIdx.x;
  if (i < B_ * H_) scale[i] = 2.0f * ALPHA_ / (sqrtf(froq[i]) * infxt[i >> 4]);
}

// ------------------------- QKV GEMM (bf16 WMMA) ----------------------------
// C[4096][3072] = xb[4096][1024] @ Wqkv + bias. Epilogue scatters into
// q[b,h,n,hd], k[b,h,n,hd], vT[b,h,hd,n] (all bf16). Double-buffered async DMA.
__global__ __launch_bounds__(128) void qkv_gemm_kernel(
    const __bf16* __restrict__ A,    // [4096][1024]
    const __bf16* __restrict__ Bt,   // [3072][1024]  (Wqkv^T)
    const float*  __restrict__ bias, // [3072]
    __bf16* __restrict__ qb, __bf16* __restrict__ kb, __bf16* __restrict__ vtb) {
  __shared__ __align__(16) __bf16 At_s[2][64][40]; // padded: row stride 80B
  __shared__ __align__(16) __bf16 Bt_s[2][64][40];
  const int tid  = threadIdx.x;
  const int lane = tid & 31, wave = tid >> 5;
  const int half = lane >> 4, l16 = lane & 15;
  const int wm = (wave >> 1) * 32, wn = (wave & 1) * 32;
  const int mbase = blockIdx.y * 64, nbase = blockIdx.x * 64;
  const int ldr = tid >> 1, ldc = (tid & 1) * 16;   // loader: 2x16B per thread

  v8f acc[2][2];
  #pragma unroll
  for (int r = 0; r < 2; ++r)
    #pragma unroll
    for (int t = 0; t < 2; ++t)
      #pragma unroll
      for (int e = 0; e < 8; ++e) acc[r][t][e] = 0.f;

  // stage tile 0
  #pragma unroll
  for (int i = 0; i < 2; ++i) {
    int c8 = ldc + i * 8;
    async_copy16(&At_s[0][ldr][c8], A  + (size_t)(mbase + ldr) * D_ + c8);
    async_copy16(&Bt_s[0][ldr][c8], Bt + (size_t)(nbase + ldr) * D_ + c8);
  }

  const int NIT = D_ / 32;
  for (int it = 0; it < NIT; ++it) {
    const int cb = it & 1;
    __syncthreads();                       // alt buffer free for overwrite
    if (it + 1 < NIT) {
      int k1 = (it + 1) * 32;
      #pragma unroll
      for (int i = 0; i < 2; ++i) {
        int c8 = ldc + i * 8;
        async_copy16(&At_s[cb ^ 1][ldr][c8], A  + (size_t)(mbase + ldr) * D_ + k1 + c8);
        async_copy16(&Bt_s[cb ^ 1][ldr][c8], Bt + (size_t)(nbase + ldr) * D_ + k1 + c8);
      }
      async_wait4();                       // tile `it` landed; tile it+1 in flight
    } else {
      async_wait0();
    }
    __syncthreads();

    FragA a[2], b[2];
    #pragma unroll
    for (int rs = 0; rs < 2; ++rs) {
      int m = wm + rs * 16 + l16;
      a[rs].q[0] = *(const uint4*)&At_s[cb][m][half * 8];
      a[rs].q[1] = *(const uint4*)&At_s[cb][m][16 + half * 8];
    }
    #pragma unroll
    for (int ts = 0; ts < 2; ++ts) {
      int n = wn + ts * 16 + l16;
      b[ts].q[0] = *(const uint4*)&Bt_s[cb][n][half * 8];
      b[ts].q[1] = *(const uint4*)&Bt_s[cb][n][16 + half * 8];
    }
    #pragma unroll
    for (int rs = 0; rs < 2; ++rs)
      #pragma unroll
      for (int ts = 0; ts < 2; ++ts)
        acc[rs][ts] = wmma_bf16(a[rs], b[ts], acc[rs][ts]);
  }

  #pragma unroll
  for (int rs = 0; rs < 2; ++rs)
    #pragma unroll
    for (int ts = 0; ts < 2; ++ts) {
      int col = nbase + wn + ts * 16 + l16;
      int which = col >> 10, cc = col & 1023;
      int h = cc >> 6, dd = cc & 63;
      float bv = bias[col];
      #pragma unroll
      for (int g = 0; g < 8; ++g) {
        int token = mbase + wm + rs * 16 + g + 8 * half;
        float val = acc[rs][ts][g] + bv;
        int b = token >> 11, n = token & (N_ - 1);
        int bh = b * H_ + h;
        __bf16 o = f2bf(val);
        if (which == 0)      qb[((size_t)bh * N_ + n) * HD_ + dd] = o;
        else if (which == 1) kb[((size_t)bh * N_ + n) * HD_ + dd] = o;
        else                 vtb[((size_t)bh * HD_ + dd) * N_ + n] = o;
      }
    }
}

// ------------------------- flash attention (bf16 WMMA) ---------------------
// One block per (b,h, 64-query block). 4 waves, each owns 16 query rows.
// Key loop in chunks of 32: S = scale*q k^T (4 wmma), online softmax,
// P through LDS, O += P V (4 wmma). k/v tiles double-buffered async DMA.
__global__ __launch_bounds__(128) void attn_kernel(
    const __bf16* __restrict__ qb, const __bf16* __restrict__ kb,
    const __bf16* __restrict__ vtb, const float* __restrict__ scale,
    __bf16* __restrict__ wgt) {
  __shared__ __align__(16) __bf16 qtile[64][72];       // [q rows][hd], padded
  __shared__ __align__(16) __bf16 ktile[2][32][72];    // [keys][hd]
  __shared__ __align__(16) __bf16 vtile[2][64][40];    // [hd][keys]
  __shared__ __align__(16) __bf16 ptile[4][16][40];    // per-wave P, [qrow][key]

  const int tid = threadIdx.x, lane = tid & 31, wave = tid >> 5;
  const int half = lane >> 4, l16 = lane & 15;
  const int qblk = blockIdx.x;            // 0..31
  const int bh   = blockIdx.y;            // 0..31
  const int h    = bh & (H_ - 1);
  const int b    = bh >> 4;
  const __bf16* qh  = qb  + (size_t)bh * N_ * HD_;
  const __bf16* kh  = kb  + (size_t)bh * N_ * HD_;
  const __bf16* vth = vtb + (size_t)bh * HD_ * N_;
  const float sc = scale[bh];

  // kv tile loader mapping (4 copies per thread per tile)
  const int kr = tid >> 2, kc = (tid & 3) * 16;   // ktile: 32 rows x 64 cols
  const int vr = tid >> 1, vc = (tid & 1) * 16;   // vtile: 64 rows x 32 cols

  // stage q tile (4 copies) then kv tile 0 (4 copies); one in-order wait
  #pragma unroll
  for (int i = 0; i < 4; ++i) {           // 512 groups: 64x64 q tile
    int g = tid * 4 + i, r = g >> 3, c8 = (g & 7) * 8;
    async_copy16(&qtile[r][c8], qh + (size_t)(qblk * 64 + r) * HD_ + c8);
  }
  #pragma unroll
  for (int i = 0; i < 2; ++i)
    async_copy16(&ktile[0][kr][kc + i * 8], kh + (size_t)kr * HD_ + kc + i * 8);
  #pragma unroll
  for (int i = 0; i < 2; ++i)
    async_copy16(&vtile[0][vr][vc + i * 8], vth + (size_t)vr * N_ + vc + i * 8);
  async_wait4();                           // q tile landed; kv tile 0 in flight
  __syncthreads();

  FragA qf[2];
  {
    int m = wave * 16 + l16;
    #pragma unroll
    for (int c = 0; c < 2; ++c) {
      int k0 = c * 32 + half * 8;
      qf[c].q[0] = *(const uint4*)&qtile[m][k0];
      qf[c].q[1] = *(const uint4*)&qtile[m][k0 + 16];
    }
  }

  float m_run[8], l_run[8];
  v8f o[4];
  #pragma unroll
  for (int g = 0; g < 8; ++g) { m_run[g] = -1e30f; l_run[g] = 0.f; }
  #pragma unroll
  for (int j = 0; j < 4; ++j)
    #pragma unroll
    for (int e = 0; e < 8; ++e) o[j][e] = 0.f;

  const int NIT = N_ / 32;
  for (int it = 0; it < NIT; ++it) {
    const int cb = it & 1;
    __syncthreads();                       // alt kv buffer free for overwrite
    if (it + 1 < NIT) {
      int kb1 = (it + 1) * 32;
      #pragma unroll
      for (int i = 0; i < 2; ++i)
        async_copy16(&ktile[cb ^ 1][kr][kc + i * 8],
                     kh + (size_t)(kb1 + kr) * HD_ + kc + i * 8);
      #pragma unroll
      for (int i = 0; i < 2; ++i)
        async_copy16(&vtile[cb ^ 1][vr][vc + i * 8],
                     vth + (size_t)vr * N_ + kb1 + vc + i * 8);
      async_wait4();                       // tile `it` landed; it+1 in flight
    } else {
      async_wait0();
    }
    __syncthreads();

    // S = sc * q k^T : two 16x16 tiles (keys 0..15, 16..31)
    v8f st[2];
    #pragma unroll
    for (int t = 0; t < 2; ++t) {
      v8f s;
      #pragma unroll
      for (int e = 0; e < 8; ++e) s[e] = 0.f;
      #pragma unroll
      for (int c = 0; c < 2; ++c) {
        FragA kf;
        int key = t * 16 + l16;
        int d0  = c * 32 + half * 8;
        kf.q[0] = *(const uint4*)&ktile[cb][key][d0];
        kf.q[1] = *(const uint4*)&ktile[cb][key][d0 + 16];
        s = wmma_bf16(qf[c], kf, s);
      }
      #pragma unroll
      for (int e = 0; e < 8; ++e) s[e] *= sc;
      st[t] = s;
    }

    // online softmax; C layout row M = g + 8*half, N = l16; reduce within half
    float fac[8];
    #pragma unroll
    for (int g = 0; g < 8; ++g) {
      float mx = fmaxf(st[0][g], st[1][g]);
      #pragma unroll
      for (int off = 1; off < 16; off <<= 1) mx = fmaxf(mx, __shfl_xor(mx, off, 32));
      float mn = fmaxf(m_run[g], mx);
      float f  = __expf(m_run[g] - mn);
      float e0 = __expf(st[0][g] - mn);
      float e1 = __expf(st[1][g] - mn);
      float ps = e0 + e1;
      #pragma unroll
      for (int off = 1; off < 16; off <<= 1) ps += __shfl_xor(ps, off, 32);
      l_run[g] = l_run[g] * f + ps;
      m_run[g] = mn;
      fac[g]   = f;
      ptile[wave][g + 8 * half][l16]      = f2bf(e0);
      ptile[wave][g + 8 * half][l16 + 16] = f2bf(e1);
    }
    __syncthreads();                       // publish ptile (cross-lane rows)

    FragA pf;
    pf.q[0] = *(const uint4*)&ptile[wave][l16][half * 8];
    pf.q[1] = *(const uint4*)&ptile[wave][l16][16 + half * 8];
    #pragma unroll
    for (int j = 0; j < 4; ++j) {
      FragA vf;
      int dcol = j * 16 + l16, kk = half * 8;
      vf.q[0] = *(const uint4*)&vtile[cb][dcol][kk];
      vf.q[1] = *(const uint4*)&vtile[cb][dcol][kk + 16];
      #pragma unroll
      for (int g = 0; g < 8; ++g) o[j][g] *= fac[g];
      o[j] = wmma_bf16(pf, vf, o[j]);
    }
  }

  #pragma unroll
  for (int j = 0; j < 4; ++j)
    #pragma unroll
    for (int g = 0; g < 8; ++g) {
      int n = qblk * 64 + wave * 16 + g + 8 * half;
      int feat = h * HD_ + j * 16 + l16;
      wgt[((size_t)(b * N_ + n)) * D_ + feat] = f2bf(o[j][g] / l_run[g]);
    }
}

// ------------------------- output projection GEMM --------------------------
__global__ __launch_bounds__(128) void proj_gemm_kernel(
    const __bf16* __restrict__ A,    // weighted [4096][1024]
    const __bf16* __restrict__ Bt,   // Wproj^T  [1024][1024]
    const float*  __restrict__ bias, // [1024]
    float* __restrict__ out) {       // [4096][1024]
  __shared__ __align__(16) __bf16 At_s[2][64][40];
  __shared__ __align__(16) __bf16 Bt_s[2][64][40];
  const int tid  = threadIdx.x;
  const int lane = tid & 31, wave = tid >> 5;
  const int half = lane >> 4, l16 = lane & 15;
  const int wm = (wave >> 1) * 32, wn = (wave & 1) * 32;
  const int mbase = blockIdx.y * 64, nbase = blockIdx.x * 64;
  const int ldr = tid >> 1, ldc = (tid & 1) * 16;

  v8f acc[2][2];
  #pragma unroll
  for (int r = 0; r < 2; ++r)
    #pragma unroll
    for (int t = 0; t < 2; ++t)
      #pragma unroll
      for (int e = 0; e < 8; ++e) acc[r][t][e] = 0.f;

  #pragma unroll
  for (int i = 0; i < 2; ++i) {
    int c8 = ldc + i * 8;
    async_copy16(&At_s[0][ldr][c8], A  + (size_t)(mbase + ldr) * D_ + c8);
    async_copy16(&Bt_s[0][ldr][c8], Bt + (size_t)(nbase + ldr) * D_ + c8);
  }

  const int NIT = D_ / 32;
  for (int it = 0; it < NIT; ++it) {
    const int cb = it & 1;
    __syncthreads();
    if (it + 1 < NIT) {
      int k1 = (it + 1) * 32;
      #pragma unroll
      for (int i = 0; i < 2; ++i) {
        int c8 = ldc + i * 8;
        async_copy16(&At_s[cb ^ 1][ldr][c8], A  + (size_t)(mbase + ldr) * D_ + k1 + c8);
        async_copy16(&Bt_s[cb ^ 1][ldr][c8], Bt + (size_t)(nbase + ldr) * D_ + k1 + c8);
      }
      async_wait4();
    } else {
      async_wait0();
    }
    __syncthreads();

    FragA a[2], b[2];
    #pragma unroll
    for (int rs = 0; rs < 2; ++rs) {
      int m = wm + rs * 16 + l16;
      a[rs].q[0] = *(const uint4*)&At_s[cb][m][half * 8];
      a[rs].q[1] = *(const uint4*)&At_s[cb][m][16 + half * 8];
    }
    #pragma unroll
    for (int ts = 0; ts < 2; ++ts) {
      int n = wn + ts * 16 + l16;
      b[ts].q[0] = *(const uint4*)&Bt_s[cb][n][half * 8];
      b[ts].q[1] = *(const uint4*)&Bt_s[cb][n][16 + half * 8];
    }
    #pragma unroll
    for (int rs = 0; rs < 2; ++rs)
      #pragma unroll
      for (int ts = 0; ts < 2; ++ts)
        acc[rs][ts] = wmma_bf16(a[rs], b[ts], acc[rs][ts]);
  }

  #pragma unroll
  for (int rs = 0; rs < 2; ++rs)
    #pragma unroll
    for (int ts = 0; ts < 2; ++ts) {
      int col = nbase + wn + ts * 16 + l16;
      float bv = bias[col];
      #pragma unroll
      for (int g = 0; g < 8; ++g) {
        int token = mbase + wm + rs * 16 + g + 8 * half;
        out[(size_t)token * D_ + col] = acc[rs][ts][g] + bv;
      }
    }
}

// ------------------------------- launch ------------------------------------

extern "C" void kernel_launch(void* const* d_in, const int* in_sizes, int n_in,
                              void* d_out, int out_size, void* d_ws, size_t ws_size,
                              hipStream_t stream) {
  const float* x     = (const float*)d_in[0];
  const float* Wqkv  = (const float*)d_in[1];
  const float* bqkv  = (const float*)d_in[2];
  const float* Wproj = (const float*)d_in[3];
  const float* bproj = (const float*)d_in[4];
  float* out = (float*)d_out;

  char* ws = (char*)d_ws;
  __bf16* xb     = (__bf16*)(ws + 0);          //  8 MB  x bf16 [4096][1024]
  __bf16* wqkvt  = (__bf16*)(ws + 8388608);    //  6 MB  Wqkv^T [3072][1024]
  __bf16* wprojt = (__bf16*)(ws + 14680064);   //  2 MB  Wproj^T [1024][1024]
  __bf16* qb     = (__bf16*)(ws + 16777216);   //  8 MB  q [32][2048][64]
  __bf16* kb     = (__bf16*)(ws + 25165824);   //  8 MB  k [32][2048][64]
  __bf16* vtb    = (__bf16*)(ws + 33554432);   //  8 MB  v^T [32][64][2048]
  __bf16* wgt    = (__bf16*)(ws + 41943040);   //  8 MB  weighted [4096][1024]
  float*  colabs = (float*)(ws + 50331648);    //  8 KB
  float*  froq   = (float*)(ws + 50339840);    //  128 B
  float*  infxt  = (float*)(ws + 50340096);    //  8 B
  float*  scl    = (float*)(ws + 50340352);    //  128 B

  f32_to_bf16_kernel<<<(B_ * N_ * D_) / 256, 256, 0, stream>>>(x, xb, B_ * N_ * D_);
  transpose_f32_to_bf16_kernel<<<(3 * D_ * D_) / 256, 256, 0, stream>>>(Wqkv, wqkvt, D_, 3 * D_);
  transpose_f32_to_bf16_kernel<<<(D_ * D_) / 256, 256, 0, stream>>>(Wproj, wprojt, D_, D_);
  colabs_kernel<<<dim3(D_ / 256, B_), 256, 0, stream>>>(x, colabs);
  infmax_kernel<<<B_, 256, 0, stream>>>(colabs, infxt);
  qkv_gemm_kernel<<<dim3((3 * D_) / 64, (B_ * N_) / 64), 128, 0, stream>>>(
      xb, wqkvt, bqkv, qb, kb, vtb);
  froq_kernel<<<B_ * H_, 256, 0, stream>>>(qb, froq);
  scale_kernel<<<1, 32, 0, stream>>>(froq, infxt, scl);
  attn_kernel<<<dim3(N_ / 64, B_ * H_), 128, 0, stream>>>(qb, kb, vtb, scl, wgt);
  proj_gemm_kernel<<<dim3(D_ / 64, (B_ * N_) / 64), 128, 0, stream>>>(
      wgt, wprojt, bproj, out);
}
